// QuantizedPolicyValueNetwork_11570641895428
// MI455X (gfx1250) — matrix operand order repaired and checked by reading, not compiled, not run on hardware
//
#include <hip/hip_runtime.h>
#include <hip/hip_bf16.h>

typedef __attribute__((ext_vector_type(16))) _Float16 v16h;
typedef __attribute__((ext_vector_type(8)))  _Float16 v8h;
typedef __attribute__((ext_vector_type(8)))  float    v8f;

#define NBOARD 19
#define NPIX   361          // 19*19
#define PADW   21           // padded image width (1-px zero ring)
#define NPPIX  441          // 21*21
#define CCH    256          // channels
#define KTOT   2304         // 9*256
#define GRPBYTES 24576      // one tap-group of B fragments: 3*8*32*16 f16
#define NTBYTES  73728      // per-ntile fragment slice: 9*8192 bytes

#if __has_builtin(__builtin_amdgcn_s_wait_asynccnt)
  #define PVN_WAIT_ASYNC(n) __builtin_amdgcn_s_wait_asynccnt(n)
#else
  #define PVN_WAIT_ASYNC(n) asm volatile("s_wait_asynccnt %0" :: "i"(n) : "memory")
#endif

// ---------------------------------------------------------------------------
// Async-DMA one contiguous 24 KB tap-group of B fragments into LDS.
// 256 threads x 6 issues x 16 B/lane  ->  ASYNCcnt += 6 per wave.
// ---------------------------------------------------------------------------
__device__ __forceinline__ void pvn_async_copy_group(const void* gsrc, uint32_t ldst)
{
    uint64_t ga = (uint64_t)(uintptr_t)gsrc + (uint32_t)(threadIdx.x * 16);
    uint32_t la = ldst + threadIdx.x * 16;
    #pragma unroll
    for (int i = 0; i < 6; ++i) {
        asm volatile("global_load_async_to_lds_b128 %0, %1, off"
                     :: "v"(la), "v"(ga) : "memory");
        la += 4096;
        ga += 4096;
    }
}

// ---------------------------------------------------------------------------
// Zero both padded activation buffers (establishes the zero halo ring).
// ---------------------------------------------------------------------------
__global__ void pvn_zero_acts(uint32_t* p, int ndw) {
    int i = blockIdx.x * 256 + threadIdx.x;
    if (i < ndw) p[i] = 0u;
}

// ---------------------------------------------------------------------------
// Per-layer weight reformat: dequantize int -> f16 and swizzle into the ISA
// 7.12.2 B-fragment order so the conv kernel can bulk-DMA it:
//   wfrag[ntile][tap][cblk][lane][j],  element = B(k = tap*256+c, n)
//   lane = (n%16) + 16*((c%16)/8), j = (c%8) + 8*((c%32)/16).
// ---------------------------------------------------------------------------
__global__ void pvn_reformat_w(const int* __restrict__ w,       // [256][256][3][3]
                               const float* __restrict__ sptr,
                               _Float16* __restrict__ wfrag)    // 589824 f16
{
    const int idx  = blockIdx.x * 256 + threadIdx.x;            // grid covers exactly
    const float s  = sptr[0];
    const int j    = idx & 15;
    const int ln   = (idx >> 4) & 31;
    const int cblk = (idx >> 9) & 7;
    const int tap  = (idx >> 12) % 9;
    const int nt   = idx / 36864;
    const int c    = cblk * 32 + ((ln >> 4) << 3) + (j & 7) + ((j >> 3) << 4);
    const int o    = nt * 16 + (ln & 15);
    wfrag[idx] = (_Float16)((float)w[o * KTOT + c * 9 + tap] * s);
}

// ---------------------------------------------------------------------------
// Init conv: 4->256 ch, 3x3, SAME. Input = 3 planes from x + ones plane.
// ---------------------------------------------------------------------------
__global__ void pvn_conv_init(const float* __restrict__ x,
                              const int* __restrict__ w,      // [256][4][3][3]
                              const float* __restrict__ sptr,
                              const float* __restrict__ bias, // [256]
                              _Float16* __restrict__ act)     // [441][256]
{
    const int n   = threadIdx.x;
    const int pix = blockIdx.x;
    const int y   = pix / NBOARD, xc = pix % NBOARD;
    const float s = sptr[0];
    float sum = 0.f;
    #pragma unroll
    for (int tap = 0; tap < 9; ++tap) {
        const int ky = tap / 3 - 1, kx = tap % 3 - 1;
        const int yy = y + ky, xx = xc + kx;
        const bool ok = (yy >= 0) & (yy < NBOARD) & (xx >= 0) & (xx < NBOARD);
        #pragma unroll
        for (int c = 0; c < 4; ++c) {
            float v = 0.f;
            if (ok) v = (c < 3) ? x[c * NPIX + yy * NBOARD + xx] : 1.0f;
            sum += v * (float)w[(n * 4 + c) * 9 + tap];
        }
    }
    float o = sum * s + bias[n];
    o = o > 0.f ? o : 0.f;
    act[((y + 1) * PADW + (xc + 1)) * CCH + n] = (_Float16)o;
}

// ---------------------------------------------------------------------------
// 3x3 conv 256->256 as implicit GEMM on v_wmma_f32_16x16x32_f16.
//   M = 361 pixels (23 tiles of 16), N = 256 out ch (16 tiles), K = 2304.
// Grid (16,3), block 256 (8 waves); each wave owns one 16x16 output tile.
// B fragments (pre-swizzled f16 in global) are streamed into LDS with
// global_load_async_to_lds_b128, double-buffered across the 3 tap groups so
// DMA of group g+1 overlaps WMMA of group g (ASYNCcnt pipelining).
// A fragments: two b128 loads/lane from the padded f16 activation image.
// Epilogue fuses bias (+ optional residual skip) + relu + f16 store.
// ---------------------------------------------------------------------------
__global__ __launch_bounds__(256) void pvn_conv3x3_wmma(
    const _Float16* __restrict__ wfrag,  // [16 nt][9 tap][8 cblk][32 ln][16 j]
    const float* __restrict__ bias,      // [256]
    const _Float16* __restrict__ actIn,  // [441][256]
    _Float16* actOut,                    // [441][256]  (may alias skip)
    const _Float16* skip)                // nullptr or [441][256]
{
    __shared__ __align__(16) _Float16 ldsB[2][3 * 8 * 32 * 16];  // 2 x 24 KB

    const int tid   = threadIdx.x;
    const int lane  = tid & 31;
    const int wave  = tid >> 5;
    const int ntile = blockIdx.x;                 // 0..15
    const int mtile = blockIdx.y * 8 + wave;      // 0..23 (23 -> idle wave)

    // A-side addressing (row of output tile this lane contributes to)
    int mrow = mtile * 16 + (lane & 15);
    if (mrow >= NPIX) mrow = NPIX - 1;            // clamp; masked at store
    const int py = mrow / NBOARD, px = mrow % NBOARD;
    const int hi = lane >> 4;                     // lane half (K-group select)

    const char*    wnt  = (const char*)wfrag + ntile * NTBYTES;
    const uint32_t lds0 = (uint32_t)(uintptr_t)(&ldsB[0][0]);   // LDS byte addr

    // prime the pipeline: group 0 -> buffer 0
    pvn_async_copy_group(wnt, lds0);

    v8f acc = {};

    for (int g = 0; g < 3; ++g) {                 // g = ky (tap row group)
        if (g < 2) {
            // kick DMA for next group into the other buffer
            pvn_async_copy_group(wnt + (g + 1) * GRPBYTES,
                                 lds0 + ((g + 1) & 1) * GRPBYTES);
            PVN_WAIT_ASYNC(6);                    // oldest 6 = current group
        } else {
            PVN_WAIT_ASYNC(0);
        }
        __syncthreads();                          // all waves' DMA visible

        if (mtile < 23) {
            const _Float16* bufB = &ldsB[g & 1][0];
            for (int t2 = 0; t2 < 3; ++t2) {
                const int pix = (py + g) * PADW + (px + t2);   // shifted pixel
                const _Float16* abase = actIn + pix * CCH + hi * 8;
                #pragma unroll
                for (int cblk = 0; cblk < 8; ++cblk) {
                    // A fragment: K = cblk*32 + {hi*8..+7, 16+hi*8..+7}
                    const v8h a0 = *(const v8h*)(abase + cblk * 32);
                    const v8h a1 = *(const v8h*)(abase + cblk * 32 + 16);
                    v16h a = __builtin_shufflevector(a0, a1,
                        0, 1, 2, 3, 4, 5, 6, 7, 8, 9, 10, 11, 12, 13, 14, 15);
                    // B fragment: 32 contiguous bytes per lane from LDS
                    const _Float16* bbase =
                        bufB + (((t2 * 8 + cblk) * 32 + lane) << 4);
                    const v8h b0 = *(const v8h*)(bbase);
                    const v8h b1 = *(const v8h*)(bbase + 8);
                    v16h b = __builtin_shufflevector(b0, b1,
                        0, 1, 2, 3, 4, 5, 6, 7, 8, 9, 10, 11, 12, 13, 14, 15);
                    acc = __builtin_amdgcn_wmma_f32_16x16x32_f16(
                        false, a, false, b, (short)0, acc, false, false);
                }
            }
        }
        __syncthreads();                          // safe to overwrite buf[g&1]
    }

    // ---- epilogue: bias (+skip) + relu + f16 store, C/D layout ----
    if (mtile < 23) {
        const int n  = ntile * 16 + (lane & 15);
        const float bn = bias[n];
        #pragma unroll
        for (int gq = 0; gq < 8; ++gq) {
            const int m = mtile * 16 + gq + hi * 8;
            if (m < NPIX) {
                const int pix = (m / NBOARD + 1) * PADW + (m % NBOARD + 1);
                float v = acc[gq] + bn;
                if (skip) v += (float)skip[pix * CCH + n];
                v = v > 0.f ? v : 0.f;
                actOut[pix * CCH + n] = (_Float16)v;
            }
        }
    }
}

// ---------------------------------------------------------------------------
// Policy/value 1x1 convs over final features (tiny).
// ---------------------------------------------------------------------------
__global__ void pvn_head_conv1x1(const _Float16* __restrict__ act,
                                 const int* __restrict__ wp, const float* sp,
                                 const float* bpv,
                                 const int* __restrict__ wv, const float* sv,
                                 const float* bvv,
                                 float* ph, float* vh)
{
    const int idx = blockIdx.x * 256 + threadIdx.x;
    if (idx < 722) {
        const int o = idx / NPIX, pix = idx % NPIX;
        const int pp = (pix / NBOARD + 1) * PADW + (pix % NBOARD + 1);
        float sum = 0.f;
        for (int c = 0; c < CCH; ++c)
            sum += (float)act[pp * CCH + c] * (float)wp[o * CCH + c];
        float v = sum * sp[0] + bpv[o];
        ph[idx] = v > 0.f ? v : 0.f;
    } else if (idx < 722 + NPIX) {
        const int pix = idx - 722;
        const int pp = (pix / NBOARD + 1) * PADW + (pix % NBOARD + 1);
        float sum = 0.f;
        for (int c = 0; c < CCH; ++c)
            sum += (float)act[pp * CCH + c] * (float)wv[c];
        float v = sum * sv[0] + bvv[0];
        vh[pix] = v > 0.f ? v : 0.f;
    }
}

// Policy FC: logits[j] = sum_i ph[i] * W[i][j] * s + bp[j]   (722 -> 361)
__global__ void pvn_policy_fc(const float* __restrict__ ph,
                              const int* __restrict__ w, const float* sptr,
                              const float* __restrict__ bp, float* out)
{
    const int j = blockIdx.x * 256 + threadIdx.x;
    if (j < NPIX) {
        float sum = 0.f;
        for (int i = 0; i < 722; ++i)
            sum += ph[i] * (float)w[i * NPIX + j];
        out[j] = sum * sptr[0] + bp[j];
    }
}

// Value FCs: 361 -> 256 (relu) -> 1 (tanh).  One block of 256 threads.
__global__ void pvn_value_fc(const float* __restrict__ vh,
                             const int* __restrict__ w1, const float* s1,
                             const float* __restrict__ b1,
                             const int* __restrict__ w2, const float* s2,
                             const float* __restrict__ b2, float* out)
{
    __shared__ float red[256];
    const int u = threadIdx.x;
    float sum = 0.f;
    for (int p = 0; p < NPIX; ++p)
        sum += vh[p] * (float)w1[p * 256 + u];
    float h = sum * s1[0] + b1[u];
    h = h > 0.f ? h : 0.f;
    red[u] = h * (float)w2[u];
    __syncthreads();
    for (int st = 128; st > 0; st >>= 1) {
        if (u < st) red[u] += red[u + st];
        __syncthreads();
    }
    if (u == 0) out[NPIX] = tanhf(red[0] * s2[0] + b2[0]);
}

// ---------------------------------------------------------------------------
extern "C" void kernel_launch(void* const* d_in, const int* in_sizes, int n_in,
                              void* d_out, int out_size, void* d_ws, size_t ws_size,
                              hipStream_t stream)
{
    const float* x       = (const float*)d_in[0];
    const int*   w_init  = (const int*)  d_in[1];
    const float* s_init  = (const float*)d_in[2];
    const float* b_init  = (const float*)d_in[3];
    const int*   rw1     = (const int*)  d_in[4];
    const float* rs1     = (const float*)d_in[5];
    const float* rb1     = (const float*)d_in[6];
    const int*   rw2     = (const int*)  d_in[7];
    const float* rs2     = (const float*)d_in[8];
    const float* rb2     = (const float*)d_in[9];
    const int*   wp      = (const int*)  d_in[10];
    const float* sp      = (const float*)d_in[11];
    const float* bpcv    = (const float*)d_in[12];
    const int*   fcpw    = (const int*)  d_in[13];
    const float* fcps    = (const float*)d_in[14];
    const float* bp      = (const float*)d_in[15];
    const int*   wv      = (const int*)  d_in[16];
    const float* sv      = (const float*)d_in[17];
    const float* bvcv    = (const float*)d_in[18];
    const int*   fv1w    = (const int*)  d_in[19];
    const float* fv1s    = (const float*)d_in[20];
    const float* bv1     = (const float*)d_in[21];
    const int*   fv2w    = (const int*)  d_in[22];
    const float* fv2s    = (const float*)d_in[23];
    const float* bv2     = (const float*)d_in[24];

    // Workspace layout (bytes):
    //   [0, 225792)        A0 padded f16 activations [441][256]
    //   [225792, 451584)   A1 padded f16 activations
    //   [458752, 1638400)  wbuf: per-layer f16 B fragments (589824 elems)
    //   [1638400, ...)     head scratch (ph 722 f32, vh 361 f32)
    _Float16* A0   = (_Float16*)d_ws;
    _Float16* A1   = A0 + NPPIX * CCH;
    _Float16* wbuf = (_Float16*)((char*)d_ws + 458752);
    float*    ph   = (float*)((char*)d_ws + 1638400);
    float*    vh   = ph + 722;
    float*    out  = (float*)d_out;

    // Zero both activation buffers (establishes zero halo rings).
    const int ndw = (2 * NPPIX * CCH * (int)sizeof(_Float16)) / 4;  // 112896
    pvn_zero_acts<<<(ndw + 255) / 256, 256, 0, stream>>>((uint32_t*)d_ws, ndw);

    // Trunk
    pvn_conv_init<<<NPIX, 256, 0, stream>>>(x, w_init, s_init, b_init, A0);
    for (int b = 0; b < 10; ++b) {
        pvn_reformat_w<<<2304, 256, 0, stream>>>(
            rw1 + (size_t)b * CCH * KTOT, rs1 + b, wbuf);
        pvn_conv3x3_wmma<<<dim3(16, 3), 256, 0, stream>>>(
            wbuf, rb1 + b * CCH, A0, A1, nullptr);

        pvn_reformat_w<<<2304, 256, 0, stream>>>(
            rw2 + (size_t)b * CCH * KTOT, rs2 + b, wbuf);
        pvn_conv3x3_wmma<<<dim3(16, 3), 256, 0, stream>>>(
            wbuf, rb2 + b * CCH, A1, A0, A0);   // fused residual add + relu
    }

    // Heads
    pvn_head_conv1x1<<<5, 256, 0, stream>>>(A0, wp, sp, bpcv, wv, sv, bvcv, ph, vh);
    pvn_policy_fc<<<2, 256, 0, stream>>>(ph, fcpw, fcps, bp, out);
    pvn_value_fc<<<1, 256, 0, stream>>>(vh, fv1w, fv1s, bv1, fv2w, fv2s, bv2, out);
}
